// OriginalDoRA_36172214567497
// MI455X (gfx1250) — matrix operand order repaired and checked by reading, compiled, not run on hardware
//
#include <hip/hip_runtime.h>

// ---------------- problem constants ----------------
#define IN_F   1024
#define OUT_F  4096
#define RNK    16
#define M_TOK  1576            // 8 * 197
#define LORA_SCALE 0.8f

// ---------------- GEMM tiling ----------------
#define BM 128
#define BN 128
#define BK 64
#define MT ((M_TOK + BM - 1) / BM)   // 13
#define MP (MT * BM)                 // 1664 (padded M)
#define LDK (BK + 8)                 // padded LDS row (bf16 units) -> conflict-free b128

typedef __attribute__((ext_vector_type(16))) __bf16 v16bf;
typedef __attribute__((ext_vector_type(8)))  float  v8f;
typedef __attribute__((ext_vector_type(4)))  int    v4i;

// ---------------- CDNA5 async global->LDS (probe-guarded) ----------------
#if defined(__has_builtin)
#  if __has_builtin(__builtin_amdgcn_global_load_async_to_lds_b128)
#    define HAVE_ASYNC_LDS 1
#  endif
#  if __has_builtin(__builtin_amdgcn_s_wait_asynccnt)
#    define HAVE_WAIT_ASYNC 1
#  endif
#endif

typedef __attribute__((address_space(1))) v4i glob_v4i;
typedef __attribute__((address_space(3))) v4i lds_v4i;

// generic LDS pointer encodes the LDS offset in its low 32 bits (aperture scheme),
// so truncation == addrspacecast generic->local
static __device__ __forceinline__ lds_v4i* to_lds(void* p) {
  return (lds_v4i*)(unsigned int)(unsigned long long)p;
}
static __device__ __forceinline__ glob_v4i* to_glob(const void* p) {
  return (glob_v4i*)(unsigned long long)p;
}

// round-to-nearest-even f32 -> bf16
static __device__ __forceinline__ unsigned short f2bf(float f) {
  unsigned int u = __float_as_uint(f);
  u += 0x7fffu + ((u >> 16) & 1u);
  return (unsigned short)(u >> 16);
}

// ============================================================
// Kernel 1: adapted_weight = (Wd + 0.8 * B@A) * magnitude -> bf16
// one block per output row n; 256 threads sweep K
// ============================================================
__global__ __launch_bounds__(256) void dora_adapt_w(
    const float* __restrict__ wd, const float* __restrict__ mag,
    const float* __restrict__ lA, const float* __restrict__ lB,
    unsigned short* __restrict__ Wb)
{
  __shared__ float Bsh[RNK];
  const int n = blockIdx.x;
  const int t = threadIdx.x;
  if (t < RNK) Bsh[t] = lB[n * RNK + t] * LORA_SCALE;
  __syncthreads();
  const float m = mag[n];
  for (int j = 0; j < IN_F; j += 256) {
    const int k = j + t;
    float acc = wd[(size_t)n * IN_F + k];
#pragma unroll
    for (int r = 0; r < RNK; ++r)
      acc = fmaf(Bsh[r], lA[r * IN_F + k], acc);
    Wb[(size_t)n * IN_F + k] = f2bf(acc * m);
  }
}

// ============================================================
// Kernel 2: x (fp32) -> bf16, zero-padded to MP rows
// ============================================================
__global__ __launch_bounds__(256) void dora_cvt_x(
    const float* __restrict__ x, unsigned short* __restrict__ Xb)
{
  const size_t i = (size_t)blockIdx.x * 256 + threadIdx.x;  // MP*IN_F total, exact grid
  const int m = (int)(i >> 10);        // IN_F == 1024
  const int k = (int)(i & 1023);
  const float v = (m < M_TOK) ? x[(size_t)m * IN_F + k] : 0.0f;
  Xb[i] = f2bf(v);
}

// ============================================================
// Kernel 3: out = Xb @ Wb^T + bias via v_wmma_f32_16x16x32_bf16
// 256 threads = 8 waves (2x4); block tile 128x128, K-step 64
// wave tile 64x32 -> 4x2 accumulators of 16x16 fp32
// ============================================================
__global__ __launch_bounds__(256) void dora_wmma_gemm(
    const unsigned short* __restrict__ Xb,
    const unsigned short* __restrict__ Wb,
    const float* __restrict__ bias,
    float* __restrict__ out)
{
  __shared__ __align__(16) unsigned short lA[BM * LDK];
  __shared__ __align__(16) unsigned short lB[BN * LDK];

  const int bm   = blockIdx.x * BM;
  const int bn   = blockIdx.y * BN;
  const int t    = threadIdx.x;
  const int wave = t >> 5;
  const int lane = t & 31;
  const int wm   = (wave >> 2) * 64;   // wave row offset in block tile
  const int wn   = (wave & 3) * 32;    // wave col offset in block tile
  const int l16  = lane & 15;
  const int khi  = lane >> 4;          // which half of the wave

  v8f acc[4][2];
#pragma unroll
  for (int mi = 0; mi < 4; ++mi)
#pragma unroll
    for (int ni = 0; ni < 2; ++ni)
      acc[mi][ni] = v8f{0.f, 0.f, 0.f, 0.f, 0.f, 0.f, 0.f, 0.f};

  for (int ks = 0; ks < IN_F; ks += BK) {
    // ---- stage A (128x64) and B (128x64) bf16 tiles into LDS ----
    // 1024 chunks of 16B per tile; 256 threads x 4 chunks
#pragma unroll
    for (int i = 0; i < 4; ++i) {
      const int c   = t + 256 * i;
      const int row = c >> 3;
      const int col = (c & 7) * 8;                    // bf16 units, 16B aligned
      const size_t ga = (size_t)(bm + row) * IN_F + ks + col;
      const size_t gb = (size_t)(bn + row) * IN_F + ks + col;
      const int la = row * LDK + col;
      const int lb = row * LDK + col;
#if HAVE_ASYNC_LDS
      __builtin_amdgcn_global_load_async_to_lds_b128(to_glob(&Xb[ga]), to_lds((void*)&lA[la]), 0, 0);
      __builtin_amdgcn_global_load_async_to_lds_b128(to_glob(&Wb[gb]), to_lds((void*)&lB[lb]), 0, 0);
#else
      *(uint4*)&lA[la] = *(const uint4*)&Xb[ga];
      *(uint4*)&lB[lb] = *(const uint4*)&Wb[gb];
#endif
    }
#if HAVE_ASYNC_LDS
#  if HAVE_WAIT_ASYNC
    __builtin_amdgcn_s_wait_asynccnt(0);
#  else
    asm volatile("s_wait_asynccnt 0" ::: "memory");
#  endif
#endif
    __syncthreads();

    // ---- compute: two K=32 WMMA slabs per K-step ----
#pragma unroll
    for (int kk = 0; kk < BK; kk += 32) {
      v16bf afrag[4];
      v16bf bfrag[2];
      // A 16x32 fragment: lanes 0-15 rows m, K {0..7,16..23}; lanes 16-31 K {8..15,24..31}
#pragma unroll
      for (int mi = 0; mi < 4; ++mi) {
        const unsigned short* p = &lA[(wm + mi * 16 + l16) * LDK + kk + khi * 8];
        ((uint4*)&afrag[mi])[0] = *(const uint4*)(p);
        ((uint4*)&afrag[mi])[1] = *(const uint4*)(p + 16);   // +16 bf16 = K+16
      }
      // B 32x16 fragment: lane holds col n; lanes 0-15 K {0..7,16..23}; lanes 16-31 K {8..15,24..31}
#pragma unroll
      for (int ni = 0; ni < 2; ++ni) {
        const unsigned short* q = &lB[(wn + ni * 16 + l16) * LDK + kk + khi * 8];
        ((uint4*)&bfrag[ni])[0] = *(const uint4*)(q);
        ((uint4*)&bfrag[ni])[1] = *(const uint4*)(q + 16);   // +16 bf16 = K+16
      }
#pragma unroll
      for (int mi = 0; mi < 4; ++mi)
#pragma unroll
        for (int ni = 0; ni < 2; ++ni)
          acc[mi][ni] = __builtin_amdgcn_wmma_f32_16x16x32_bf16(
              false, afrag[mi], false, bfrag[ni], (short)0, acc[mi][ni], false, false);
    }
    __syncthreads();
  }

  // ---- epilogue: C/D layout (VGPR v: lanes 0-15 -> M=v, lanes 16-31 -> M=v+8) ----
#pragma unroll
  for (int ni = 0; ni < 2; ++ni) {
    const int n  = bn + wn + ni * 16 + l16;
    const float bv = bias[n];
#pragma unroll
    for (int mi = 0; mi < 4; ++mi) {
      const int mbase = bm + wm + mi * 16 + khi * 8;
#pragma unroll
      for (int v = 0; v < 8; ++v) {
        const int m = mbase + v;
        if (m < M_TOK)
          out[(size_t)m * OUT_F + n] = acc[mi][ni][v] + bv;
      }
    }
  }
}

// ============================================================
extern "C" void kernel_launch(void* const* d_in, const int* in_sizes, int n_in,
                              void* d_out, int out_size, void* d_ws, size_t ws_size,
                              hipStream_t stream) {
  (void)in_sizes; (void)n_in; (void)out_size; (void)ws_size;
  const float* x    = (const float*)d_in[0];
  const float* wd   = (const float*)d_in[1];
  const float* mag  = (const float*)d_in[2];
  const float* lA   = (const float*)d_in[3];
  const float* lB   = (const float*)d_in[4];
  const float* bias = (const float*)d_in[5];
  float* out = (float*)d_out;

  // workspace: Wb bf16 [OUT_F][IN_F] then Xb bf16 [MP][IN_F]  (~11.8 MB)
  unsigned short* Wb = (unsigned short*)d_ws;
  unsigned short* Xb = Wb + (size_t)OUT_F * IN_F;

  dora_adapt_w<<<OUT_F, 256, 0, stream>>>(wd, mag, lA, lB, Wb);
  dora_cvt_x<<<(MP * IN_F) / 256, 256, 0, stream>>>(x, Xb);
  dora_wmma_gemm<<<dim3(MT, OUT_F / BN), 256, 0, stream>>>(Xb, Wb, bias, out);
}
